// ModalAttention_88536455840318
// MI455X (gfx1250) — matrix operand order repaired and checked by reading, not compiled, not run on hardware
//
#include <hip/hip_runtime.h>

// ---------------------------------------------------------------------------
// Modal attention for MI455X (gfx1250): bf16 WMMA everywhere, fp32 accumulate.
// B=2, T=2048, C=1024, H=16, D=64, N_WORLDS=4.
// Round 2: 32-row wave tiles in the projection GEMMs (halves B traffic per
// WMMA), and async LDS staging (global_load_async_to_lds_b128 / ASYNCcnt)
// for the V tile in the attention kernel.
// ---------------------------------------------------------------------------

#define BATCH   2
#define SEQ     2048
#define CDIM    1024
#define NHEAD   16
#define HDIM    64
#define MROWS   (BATCH * SEQ)          // 4096

typedef __bf16 bf16_t;
typedef __attribute__((ext_vector_type(16))) __bf16 v16bf;
typedef __attribute__((ext_vector_type(8)))  __bf16 v8bf;
typedef __attribute__((ext_vector_type(8)))  float  v8f;

__device__ inline v8f wmma_bf16(v16bf a, v16bf b, v8f c) {
  // D = A(16x32 bf16) * B(32x16 bf16) + C(16x16 f32)
  return __builtin_amdgcn_wmma_f32_16x16x32_bf16(
      /*neg_a=*/false, a, /*neg_b=*/false, b,
      /*c_mod=*/(short)0, c, /*reuse_a=*/false, /*reuse_b=*/false);
}

__device__ inline bf16_t f2bf(float f) {  // round-to-nearest-even f32 -> bf16
  union { float f; unsigned u; } in; in.f = f;
  unsigned u = in.u;
  u += 0x7FFFu + ((u >> 16) & 1u);
  unsigned short s = (unsigned short)(u >> 16);
  union { unsigned short s; bf16_t b; } out; out.s = s;
  return out.b;
}

// A fragment, 16(M) x 32(K) bf16, row-major source with stride lda.
// ISA layout: lane<16 -> m=lane, halves 0..7 = K0..7, halves 8..15 = K16..23;
//             lane>=16 -> m=lane-16, K8..15 / K24..31.
__device__ inline v16bf load_a16x32(const bf16_t* A, int lda) {
  const int lane = threadIdx.x & 31;
  const int m  = lane & 15;
  const int kb = (lane >> 4) << 3;        // 0 or 8
  const bf16_t* p = A + (long)m * lda + kb;
  v8bf lo = *(const v8bf*)(p);            // K = kb .. kb+7
  v8bf hi = *(const v8bf*)(p + 16);       // K = kb+16 .. kb+23
  v16bf r;
#pragma unroll
  for (int i = 0; i < 8; ++i) { r[i] = lo[i]; r[i + 8] = hi[i]; }
  return r;
}

// B fragment, 32(K) x 16(N) bf16, row-major source with stride ldb.
// Lane l holds column n=l%16; halves 0..7 = K kb..kb+7, 8..15 = K kb+16..kb+23 (kb=8*(l/16)).
__device__ inline v16bf load_b32x16(const bf16_t* Bp, int ldb) {
  const int lane = threadIdx.x & 31;
  const int n  = lane & 15;
  const int kb = (lane >> 4) << 3;
  v16bf r;
#pragma unroll
  for (int i = 0; i < 8; ++i) {
    r[i]     = Bp[(long)(kb + i) * ldb + n];
    r[i + 8] = Bp[(long)(kb + 16 + i) * ldb + n];
  }
  return r;
}

// ---------------------------------------------------------------------------
// Kernel 1: fp32 -> bf16 convert; weights also transposed (Wt[k*C+n]=W[n*C+k])
// ---------------------------------------------------------------------------
__global__ void cvt_kernel(const float* __restrict__ x,
                           const float* __restrict__ Wq, const float* __restrict__ Wk,
                           const float* __restrict__ Wv, const float* __restrict__ Wp,
                           bf16_t* __restrict__ xb,
                           bf16_t* __restrict__ wqt, bf16_t* __restrict__ wkt,
                           bf16_t* __restrict__ wvt, bf16_t* __restrict__ wpt) {
  const long NX = (long)MROWS * CDIM;
  long i = (long)blockIdx.x * blockDim.x + threadIdx.x;
  if (i < NX) { xb[i] = f2bf(x[i]); return; }
  i -= NX;
  const long WSZ = (long)CDIM * CDIM;
  if (i >= 4 * WSZ) return;
  int  w = (int)(i / WSZ);
  long j = i % WSZ;
  int  kk = (int)(j / CDIM), n = (int)(j % CDIM);
  const float* W  = (w == 0) ? Wq : (w == 1) ? Wk : (w == 2) ? Wv : Wp;
  bf16_t*      Wt = (w == 0) ? wqt : (w == 1) ? wkt : (w == 2) ? wvt : wpt;
  Wt[j] = f2bf(W[(long)n * CDIM + kk]);
}

// ---------------------------------------------------------------------------
// Kernel 2: q,k,v = xb @ Wt  (y = x W^T). grid(16, 16(=head), 3), block 256.
// Each wave computes a 32x64 tile (2 A-frags share every B-frag).
// q scaled by 1/sqrt(64). Out layout: [B,H,T,D] bf16.
// ---------------------------------------------------------------------------
__global__ void qkv_kernel(const bf16_t* __restrict__ xb,
                           const bf16_t* __restrict__ wqt, const bf16_t* __restrict__ wkt,
                           const bf16_t* __restrict__ wvt,
                           bf16_t* __restrict__ q, bf16_t* __restrict__ k,
                           bf16_t* __restrict__ v) {
  const int wave = threadIdx.x >> 5;
  const int m0 = blockIdx.x * 256 + wave * 32;
  const int h  = blockIdx.y;
  const int n0 = h * HDIM;
  const int which = blockIdx.z;
  const bf16_t* Wt  = (which == 0) ? wqt : (which == 1) ? wkt : wvt;
  bf16_t*       out = (which == 0) ? q   : (which == 1) ? k   : v;

  v8f acc[2][4];
#pragma unroll
  for (int u = 0; u < 2; ++u)
#pragma unroll
    for (int t = 0; t < 4; ++t)
      acc[u][t] = (v8f){0.f,0.f,0.f,0.f,0.f,0.f,0.f,0.f};

  for (int kk = 0; kk < CDIM; kk += 32) {
    v16bf a0 = load_a16x32(xb + (long)m0 * CDIM + kk, CDIM);
    v16bf a1 = load_a16x32(xb + (long)(m0 + 16) * CDIM + kk, CDIM);
#pragma unroll
    for (int t = 0; t < 4; ++t) {
      v16bf b = load_b32x16(Wt + (long)kk * CDIM + n0 + t * 16, CDIM);
      acc[0][t] = wmma_bf16(a0, b, acc[0][t]);
      acc[1][t] = wmma_bf16(a1, b, acc[1][t]);
    }
  }
  const float scale = (which == 0) ? 0.125f : 1.0f;   // 1/sqrt(HDIM) folded into q
  const int lane = threadIdx.x & 31;
  const int nn = lane & 15, mb = (lane >> 4) << 3;
#pragma unroll
  for (int u = 0; u < 2; ++u)
#pragma unroll
    for (int t = 0; t < 4; ++t)
#pragma unroll
      for (int r = 0; r < 8; ++r) {
        const int m  = m0 + u * 16 + mb + r;
        const int bb = m >> 11;          // / SEQ
        const int tt = m & (SEQ - 1);
        out[(((long)(bb * NHEAD + h)) * SEQ + tt) * HDIM + t * 16 + nn] =
            f2bf(acc[u][t][r] * scale);
      }
}

// ---------------------------------------------------------------------------
// Kernel 3: flash attention with Kripke modal mask. grid(16, 32(=b*h)), 256.
// Per wave: 16 q rows, streaming over 64-key tiles.
//   K tile : staged transposed through LDS (ds stores, layout change needed)
//   V tile : staged via gfx1250 async DMA (global_load_async_to_lds_b128,
//            ASYNCcnt) -- byte-identical copy, no transform needed.
// Mask is loop-invariant per fragment element: qw = r%4, kw = lane%4.
// ---------------------------------------------------------------------------
__global__ void attn_kernel(const bf16_t* __restrict__ q, const bf16_t* __restrict__ k,
                            const bf16_t* __restrict__ v, const int* __restrict__ accm,
                            bf16_t* __restrict__ o) {
  __shared__ bf16_t Kt[HDIM * 64];          // [dim][key]   8 KB (transposed)
  __shared__ bf16_t Vt[64 * HDIM];          // [key][dim]   8 KB (async copy)
  __shared__ bf16_t Pb[8 * 16 * 64];        // per-wave P   16 KB

  const int wave = threadIdx.x >> 5;
  const int lane = threadIdx.x & 31;
  const int bh = blockIdx.y;
  const int b  = bh >> 4, h = bh & 15;
  const bf16_t* qb = q + (long)bh * SEQ * HDIM;
  const bf16_t* kb = k + (long)bh * SEQ * HDIM;
  const bf16_t* vb = v + (long)bh * SEQ * HDIM;
  const int q0 = blockIdx.x * 128 + wave * 16;
  bf16_t* Pw = Pb + wave * 16 * 64;

  const int nn = lane & 15, mb = (lane >> 4) << 3;

  // Loop-invariant modal mask bits: q-world = (q0+mb+r)%4 = r%4 ; k-world = lane%4
  const int kw = lane & 3;
  bool mask[8];
#pragma unroll
  for (int r = 0; r < 8; ++r) mask[r] = (accm[(r & 3) * 4 + kw] != 0);

  // Q fragments are loop-invariant: hoist.
  v16bf aq[2];
#pragma unroll
  for (int ks = 0; ks < 2; ++ks)
    aq[ks] = load_a16x32(qb + (long)q0 * HDIM + ks * 32, HDIM);

  v8f Oacc[4];
#pragma unroll
  for (int t = 0; t < 4; ++t) Oacc[t] = (v8f){0.f,0.f,0.f,0.f,0.f,0.f,0.f,0.f};
  float mrow[8], lrow[8];
#pragma unroll
  for (int r = 0; r < 8; ++r) { mrow[r] = -3.0e38f; lrow[r] = 0.f; }

  const unsigned vt_lds = (unsigned)(unsigned long long)(void*)Vt; // LDS byte offset

  for (int kv0 = 0; kv0 < SEQ; kv0 += 64) {
    // --- V tile: async DMA global -> LDS (8 KB = 2 x 256 lanes x 16 B) -----
    {
      const char* gsrc = (const char*)(vb + (long)kv0 * HDIM);
      const int byte0 = threadIdx.x * 16;
#pragma unroll
      for (int p = 0; p < 2; ++p) {
        unsigned laddr = vt_lds + byte0 + p * 4096;
        const char* gaddr = gsrc + byte0 + p * 4096;
        asm volatile("global_load_async_to_lds_b128 %0, %1, off"
                     :: "v"(laddr), "v"(gaddr) : "memory");
      }
    }
    // --- K tile: cooperative transposed stage: Kt[dim*64+key] = kb[key][dim]
    for (int i = threadIdx.x; i < 64 * 64; i += 256) {
      const int ky = i >> 6, dx = i & 63;
      Kt[dx * 64 + ky] = kb[(long)(kv0 + ky) * HDIM + dx];
    }
    if (kv0 + 64 < SEQ)   // gfx1250 global_prefetch of next K tile
      __builtin_prefetch(kb + (long)(kv0 + 64) * HDIM + lane * 8, 0, 1);
    asm volatile("s_wait_asynccnt 0x0" ::: "memory");   // V DMA landed
    __syncthreads();                                    // publish Kt + Vt

    // S = Q * K^T  (scale already folded into q)
    v8f s[4];
#pragma unroll
    for (int t = 0; t < 4; ++t) s[t] = (v8f){0.f,0.f,0.f,0.f,0.f,0.f,0.f,0.f};
#pragma unroll
    for (int ks = 0; ks < 2; ++ks) {
#pragma unroll
      for (int t = 0; t < 4; ++t) {
        v16bf bk = load_b32x16(&Kt[(ks * 32) * 64 + t * 16], 64);
        s[t] = wmma_bf16(aq[ks], bk, s[t]);
      }
    }

    // Online softmax per row (rows live across 16-lane halves in D layout)
#pragma unroll
    for (int r = 0; r < 8; ++r) {
      float smax = -3.0e38f;
#pragma unroll
      for (int t = 0; t < 4; ++t) {
        const float sv = mask[r] ? s[t][r] : -3.0e38f;
        s[t][r] = sv;
        smax = fmaxf(smax, sv);
      }
#pragma unroll
      for (int off = 1; off < 16; off <<= 1)
        smax = fmaxf(smax, __shfl_xor(smax, off, 32));
      const float mnew = fmaxf(mrow[r], smax);
      const float corr = __expf(mrow[r] - mnew);
      mrow[r] = mnew;
      float psum = 0.f;
#pragma unroll
      for (int t = 0; t < 4; ++t) {
        const float p = __expf(s[t][r] - mnew);
        s[t][r] = p;
        psum += p;
      }
#pragma unroll
      for (int off = 1; off < 16; off <<= 1)
        psum += __shfl_xor(psum, off, 32);
      lrow[r] = lrow[r] * corr + psum;
#pragma unroll
      for (int t = 0; t < 4; ++t) Oacc[t][r] *= corr;
    }

    // P (D layout, f32) -> LDS bf16 row-major, so it reloads as A fragments
#pragma unroll
    for (int t = 0; t < 4; ++t)
#pragma unroll
      for (int r = 0; r < 8; ++r)
        Pw[(mb + r) * 64 + t * 16 + nn] = f2bf(s[t][r]);

    // O += P * V   (V B-fragments now come from LDS)
#pragma unroll
    for (int ks = 0; ks < 2; ++ks) {
      v16bf ap = load_a16x32(Pw + ks * 32, 64);
#pragma unroll
      for (int t = 0; t < 4; ++t) {
        v16bf bv = load_b32x16(&Vt[(ks * 32) * HDIM + t * 16], HDIM);
        Oacc[t] = wmma_bf16(ap, bv, Oacc[t]);
      }
    }
    __syncthreads();   // protect Kt/Vt before next tile overwrites them
  }

  // Normalize and scatter to o[B*T, C] (merged-head layout) as bf16
#pragma unroll
  for (int r = 0; r < 8; ++r) {
    const float inv = 1.0f / lrow[r];
#pragma unroll
    for (int t = 0; t < 4; ++t) Oacc[t][r] *= inv;
  }
#pragma unroll
  for (int t = 0; t < 4; ++t)
#pragma unroll
    for (int r = 0; r < 8; ++r) {
      const int m = q0 + mb + r;
      o[((long)(b * SEQ + m)) * CDIM + h * HDIM + t * 16 + nn] = f2bf(Oacc[t][r]);
    }
}

// ---------------------------------------------------------------------------
// Kernel 4: out = o @ Wproj^T, fp32 output. grid(16, 16), block 256.
// Each wave computes a 32x64 tile.
// ---------------------------------------------------------------------------
__global__ void proj_kernel(const bf16_t* __restrict__ ob,
                            const bf16_t* __restrict__ wpt,
                            float* __restrict__ out) {
  const int wave = threadIdx.x >> 5;
  const int m0 = blockIdx.x * 256 + wave * 32;
  const int n0 = blockIdx.y * 64;
  v8f acc[2][4];
#pragma unroll
  for (int u = 0; u < 2; ++u)
#pragma unroll
    for (int t = 0; t < 4; ++t)
      acc[u][t] = (v8f){0.f,0.f,0.f,0.f,0.f,0.f,0.f,0.f};
  for (int kk = 0; kk < CDIM; kk += 32) {
    v16bf a0 = load_a16x32(ob + (long)m0 * CDIM + kk, CDIM);
    v16bf a1 = load_a16x32(ob + (long)(m0 + 16) * CDIM + kk, CDIM);
#pragma unroll
    for (int t = 0; t < 4; ++t) {
      v16bf b = load_b32x16(wpt + (long)kk * CDIM + n0 + t * 16, CDIM);
      acc[0][t] = wmma_bf16(a0, b, acc[0][t]);
      acc[1][t] = wmma_bf16(a1, b, acc[1][t]);
    }
  }
  const int lane = threadIdx.x & 31;
  const int nn = lane & 15, mb = (lane >> 4) << 3;
#pragma unroll
  for (int u = 0; u < 2; ++u)
#pragma unroll
    for (int t = 0; t < 4; ++t)
#pragma unroll
      for (int r = 0; r < 8; ++r)
        out[(long)(m0 + u * 16 + mb + r) * CDIM + n0 + t * 16 + nn] = acc[u][t][r];
}

// ---------------------------------------------------------------------------
extern "C" void kernel_launch(void* const* d_in, const int* in_sizes, int n_in,
                              void* d_out, int out_size, void* d_ws, size_t ws_size,
                              hipStream_t stream) {
  const float* x  = (const float*)d_in[0];
  const float* Wq = (const float*)d_in[1];
  const float* Wk = (const float*)d_in[2];
  const float* Wv = (const float*)d_in[3];
  const float* Wp = (const float*)d_in[4];
  const int*   am = (const int*)d_in[5];
  float* out = (float*)d_out;

  char* ws = (char*)d_ws;
  const size_t MB = 1024ull * 1024ull;
  bf16_t* xb   = (bf16_t*)(ws + 0);          // 8 MB  [B*T, C]
  bf16_t* wqt  = (bf16_t*)(ws + 8  * MB);    // 2 MB  [C, C] transposed
  bf16_t* wkt  = (bf16_t*)(ws + 10 * MB);
  bf16_t* wvt  = (bf16_t*)(ws + 12 * MB);
  bf16_t* wpt  = (bf16_t*)(ws + 14 * MB);
  bf16_t* qbuf = (bf16_t*)(ws + 16 * MB);    // 8 MB  [B,H,T,D]
  bf16_t* kbuf = (bf16_t*)(ws + 24 * MB);
  bf16_t* vbuf = (bf16_t*)(ws + 32 * MB);
  bf16_t* obuf = (bf16_t*)(ws + 40 * MB);    // 8 MB  [B*T, C]

  // 1) convert + transpose weights
  {
    const long total = (long)MROWS * CDIM + 4L * CDIM * CDIM;  // 8,388,608
    const int blocks = (int)((total + 255) / 256);
    cvt_kernel<<<blocks, 256, 0, stream>>>(x, Wq, Wk, Wv, Wp,
                                           xb, wqt, wkt, wvt, wpt);
  }
  // 2) q,k,v projections (z selects which weight)
  qkv_kernel<<<dim3(MROWS / 256, NHEAD, 3), 256, 0, stream>>>(
      xb, wqt, wkt, wvt, qbuf, kbuf, vbuf);
  // 3) masked flash attention (async V staging + LDS-transposed K)
  attn_kernel<<<dim3(SEQ / 128, BATCH * NHEAD), 256, 0, stream>>>(
      qbuf, kbuf, vbuf, am, obuf);
  // 4) output projection (fp32 result)
  proj_kernel<<<dim3(MROWS / 256, CDIM / 64), 256, 0, stream>>>(obuf, wpt, out);
}
